// DGCNN_34694745817252
// MI455X (gfx1250) — matrix operand order repaired and checked by reading, not compiled, not run on hardware
//
#include <hip/hip_runtime.h>
#include <hip/hip_bf16.h>

typedef __attribute__((ext_vector_type(16))) __bf16 v16bf;
typedef __attribute__((ext_vector_type(8)))  __bf16 v8bf;
typedef __attribute__((ext_vector_type(8)))  float  v8f;

#define NPTS  1024
#define BATCH 4
#define NROWS (BATCH * NPTS)   // 4096
#define KNN   20
#define CATCH 6400
#define LDSPAD 40              // 32 bf16 payload + 8 pad -> 80B row, conflict-free b128 reads
#define LRELU(v) ((v) > 0.f ? (v) : 0.2f * (v))

__device__ __forceinline__ __bf16 to_bf(float f) { return (__bf16)f; }

__device__ __forceinline__ v8bf cvt8(float4 a, float4 b) {
    v8bf p;
    p[0] = (__bf16)a.x; p[1] = (__bf16)a.y; p[2] = (__bf16)a.z; p[3] = (__bf16)a.w;
    p[4] = (__bf16)b.x; p[5] = (__bf16)b.y; p[6] = (__bf16)b.z; p[7] = (__bf16)b.w;
    return p;
}
__device__ __forceinline__ v8bf cvt8_diff(float4 a, float4 b, float4 c, float4 d) {
    v8bf p;  // (c,d) - (a,b)
    p[0] = (__bf16)(c.x - a.x); p[1] = (__bf16)(c.y - a.y);
    p[2] = (__bf16)(c.z - a.z); p[3] = (__bf16)(c.w - a.w);
    p[4] = (__bf16)(d.x - b.x); p[5] = (__bf16)(d.y - b.y);
    p[6] = (__bf16)(d.z - b.z); p[7] = (__bf16)(d.w - b.w);
    return p;
}

// ---------------------------------------------------------------------------
// transpose x (B,3,N) -> h (B*N, 3)
__global__ void k_transpose(const float* __restrict__ x, float* __restrict__ h) {
    int t = blockIdx.x * 256 + threadIdx.x;
    if (t >= NROWS) return;
    int b = t >> 10, n = t & 1023;
    for (int c = 0; c < 3; ++c)
        h[t * 3 + c] = x[(b * 3 + c) * NPTS + n];
}

// squared norms per point
__global__ void k_sqnorm(const float* __restrict__ h, int C, float* __restrict__ xx) {
    int t = blockIdx.x * 256 + threadIdx.x;
    if (t >= NROWS) return;
    float s = 0.f;
    for (int c = 0; c < C; ++c) { float v = h[(size_t)t * C + c]; s += v * v; }
    xx[t] = s;
}

// ---------------------------------------------------------------------------
// Gram / distance GEMM: dist[b,n,m] = 2*<h_n,h_m> - xx_n - xx_m  (bf16 WMMA)
template<bool FULLK>
__global__ __launch_bounds__(256)
void k_wmma_gram(const float* __restrict__ H, int Kdim,
                 const float* __restrict__ xx, float* __restrict__ dist) {
    __shared__ __bf16 As[64][LDSPAD];
    __shared__ __bf16 Bs[64][LDSPAD];
    const int b    = blockIdx.z;
    const float* Hb = H + (size_t)b * NPTS * Kdim;
    const int tid  = threadIdx.x;
    const int lane = tid & 31, wave = tid >> 5;
    const int rw = wave >> 1, wc = wave & 1;
    const int row0 = blockIdx.x * 64, col0 = blockIdx.y * 64;
    v8f acc0 = {}, acc1 = {};
    const int lr = tid >> 2;            // 0..63 tile row
    const int lk = (tid & 3) * 8;       // 0,8,16,24

    for (int k0 = 0; k0 < Kdim; k0 += 32) {
        if (FULLK) {
            const float4* ap = (const float4*)(Hb + (size_t)(row0 + lr) * Kdim + k0 + lk);
            const float4* bp = (const float4*)(Hb + (size_t)(col0 + lr) * Kdim + k0 + lk);
            float4 a0 = ap[0], a1 = ap[1];
            float4 c0 = bp[0], c1 = bp[1];
            *(v8bf*)&As[lr][lk] = cvt8(a0, a1);
            *(v8bf*)&Bs[lr][lk] = cvt8(c0, c1);
            if (k0 + 32 < Kdim) {
                __builtin_prefetch(Hb + (size_t)(row0 + lr) * Kdim + k0 + 32 + lk, 0, 1);
                __builtin_prefetch(Hb + (size_t)(col0 + lr) * Kdim + k0 + 32 + lk, 0, 1);
            }
        } else {
#pragma unroll
            for (int u = 0; u < 8; ++u) {
                int k = lk + u, kk = k0 + k;
                As[lr][k] = to_bf(kk < Kdim ? Hb[(size_t)(row0 + lr) * Kdim + kk] : 0.f);
                Bs[lr][k] = to_bf(kk < Kdim ? Hb[(size_t)(col0 + lr) * Kdim + kk] : 0.f);
            }
        }
        __syncthreads();
        // A fragment: lane<16 -> M=lane, K {0..15}; lane>=16 -> K {16..31}
        v16bf af;
        const int m  = rw * 16 + (lane & 15);
        const int kb = (lane >> 4) * 16;
#pragma unroll
        for (int v = 0; v < 8; ++v) {
            int k = kb + 2 * v;
            af[2 * v] = As[m][k]; af[2 * v + 1] = As[m][k + 1];
        }
        const int kb2 = (lane >> 4) * 16;
#pragma unroll
        for (int t = 0; t < 2; ++t) {
            const int n = wc * 32 + t * 16 + (lane & 15);
            v16bf bfr;
#pragma unroll
            for (int j = 0; j < 8; ++j) {
                bfr[2 * j] = Bs[n][kb2 + 2 * j]; bfr[2 * j + 1] = Bs[n][kb2 + 2 * j + 1];
            }
            if (t == 0) acc0 = __builtin_amdgcn_wmma_f32_16x16x32_bf16(false, af, false, bfr, (short)0, acc0, false, false);
            else        acc1 = __builtin_amdgcn_wmma_f32_16x16x32_bf16(false, af, false, bfr, (short)0, acc1, false, false);
        }
        __syncthreads();
    }
    const int nlo = lane & 15, mhi = (lane >> 4) * 8;
#pragma unroll
    for (int r = 0; r < 8; ++r) {
        int row = row0 + rw * 16 + mhi + r;
        int col = col0 + wc * 32 + nlo;
        float xr = xx[b * NPTS + row];
        size_t base = ((size_t)b * NPTS + row) * NPTS;
        dist[base + col]      = 2.f * acc0[r] - xr - xx[b * NPTS + col];
        dist[base + col + 16] = 2.f * acc1[r] - xr - xx[b * NPTS + col + 16];
    }
}

// ---------------------------------------------------------------------------
// top-20 per row (one wave32 per row), tie-break: smaller column index
__global__ __launch_bounds__(256)
void k_topk(const float* __restrict__ dist, int* __restrict__ idx) {
    const int wave = threadIdx.x >> 5, lane = threadIdx.x & 31;
    const int row = blockIdx.x * 8 + wave;
    const float* d = dist + (size_t)row * NPTS;
    unsigned chosen = 0;
    for (int it = 0; it < KNN; ++it) {
        float bv = -3.4e38f; int bc = 1 << 30;
        for (int j = 0; j < 32; ++j) {
            if ((chosen >> j) & 1) continue;
            float v = d[j * 32 + lane];
            if (v > bv) { bv = v; bc = j * 32 + lane; }
        }
        for (int off = 16; off > 0; off >>= 1) {
            float ov = __shfl_down(bv, off, 32);
            int   oc = __shfl_down(bc, off, 32);
            if (ov > bv || (ov == bv && oc < bc)) { bv = ov; bc = oc; }
        }
        bc = __shfl(bc, 0, 32);
        if ((bc & 31) == lane) chosen |= 1u << (bc >> 5);
        if (lane == 0) idx[row * KNN + it] = bc;
    }
}

// ---------------------------------------------------------------------------
// C = A (M x K, row-major) * W^T   with bf16 WMMA.
// DUAL=1 (edge conv): Y1 = A*W1^T, Y2 = A*(W2-W1)^T with W = [W1|W2], ldw = 2K.
// DUAL=0: Y1 = A*W^T, ldw = K.
template<int DUAL, bool FULLK>
__global__ __launch_bounds__(256)
void k_wmma_gemm_nt(const float* __restrict__ A, int Kdim,
                    const float* __restrict__ W, int ldw,
                    float* __restrict__ Y1, float* __restrict__ Y2, int Ocols) {
    __shared__ __bf16 As[64][LDSPAD];
    __shared__ __bf16 Bs1[64][LDSPAD];
    __shared__ __bf16 Bs2[DUAL ? 64 : 1][LDSPAD];
    const int tid  = threadIdx.x;
    const int lane = tid & 31, wave = tid >> 5;
    const int rw = wave >> 1, wc = wave & 1;
    const int row0 = blockIdx.x * 64, col0 = blockIdx.y * 64;
    v8f a00 = {}, a01 = {}, a10 = {}, a11 = {};
    const int lr = tid >> 2;
    const int lk = (tid & 3) * 8;

    for (int k0 = 0; k0 < Kdim; k0 += 32) {
        if (FULLK) {
            const float4* ap = (const float4*)(A + (size_t)(row0 + lr) * Kdim + k0 + lk);
            float4 a0 = ap[0], a1 = ap[1];
            *(v8bf*)&As[lr][lk] = cvt8(a0, a1);
            const float4* wp = (const float4*)(W + (size_t)(col0 + lr) * ldw + k0 + lk);
            float4 w0 = wp[0], w1 = wp[1];
            *(v8bf*)&Bs1[lr][lk] = cvt8(w0, w1);
            if (DUAL) {
                const float4* wq = (const float4*)(W + (size_t)(col0 + lr) * ldw + Kdim + k0 + lk);
                float4 u0 = wq[0], u1 = wq[1];
                *(v8bf*)&Bs2[lr][lk] = cvt8_diff(w0, w1, u0, u1);
            }
            if (k0 + 32 < Kdim) {
                __builtin_prefetch(A + (size_t)(row0 + lr) * Kdim + k0 + 32 + lk, 0, 1);
                __builtin_prefetch(W + (size_t)(col0 + lr) * ldw + k0 + 32 + lk, 0, 1);
                if (DUAL)
                    __builtin_prefetch(W + (size_t)(col0 + lr) * ldw + Kdim + k0 + 32 + lk, 0, 1);
            }
        } else {
#pragma unroll
            for (int u = 0; u < 8; ++u) {
                int k = lk + u, kk = k0 + k;
                As[lr][k] = to_bf(kk < Kdim ? A[(size_t)(row0 + lr) * Kdim + kk] : 0.f);
                float w1 = kk < Kdim ? W[(size_t)(col0 + lr) * ldw + kk] : 0.f;
                Bs1[lr][k] = to_bf(w1);
                if (DUAL) {
                    float w2 = kk < Kdim ? W[(size_t)(col0 + lr) * ldw + Kdim + kk] : 0.f;
                    Bs2[lr][k] = to_bf(w2 - w1);
                }
            }
        }
        __syncthreads();
        v16bf af;
        const int m  = rw * 16 + (lane & 15);
        const int kb = (lane >> 4) * 16;
#pragma unroll
        for (int v = 0; v < 8; ++v) {
            int k = kb + 2 * v;
            af[2 * v] = As[m][k]; af[2 * v + 1] = As[m][k + 1];
        }
        const int kb2 = (lane >> 4) * 16;
#pragma unroll
        for (int t = 0; t < 2; ++t) {
            const int n = wc * 32 + t * 16 + (lane & 15);
            v16bf b1;
#pragma unroll
            for (int j = 0; j < 8; ++j) {
                b1[2 * j] = Bs1[n][kb2 + 2 * j]; b1[2 * j + 1] = Bs1[n][kb2 + 2 * j + 1];
            }
            if (t == 0) a00 = __builtin_amdgcn_wmma_f32_16x16x32_bf16(false, af, false, b1, (short)0, a00, false, false);
            else        a01 = __builtin_amdgcn_wmma_f32_16x16x32_bf16(false, af, false, b1, (short)0, a01, false, false);
            if (DUAL) {
                v16bf b2;
#pragma unroll
                for (int j = 0; j < 8; ++j) {
                    b2[2 * j] = Bs2[n][kb2 + 2 * j]; b2[2 * j + 1] = Bs2[n][kb2 + 2 * j + 1];
                }
                if (t == 0) a10 = __builtin_amdgcn_wmma_f32_16x16x32_bf16(false, af, false, b2, (short)0, a10, false, false);
                else        a11 = __builtin_amdgcn_wmma_f32_16x16x32_bf16(false, af, false, b2, (short)0, a11, false, false);
            }
        }
        __syncthreads();
    }
    const int nlo = lane & 15, mhi = (lane >> 4) * 8;
#pragma unroll
    for (int r = 0; r < 8; ++r) {
        size_t row = (size_t)(row0 + rw * 16 + mhi + r);
        int c0 = col0 + wc * 32 + nlo;
        Y1[row * Ocols + c0]      = a00[r];
        Y1[row * Ocols + c0 + 16] = a01[r];
        if (DUAL) {
            Y2[row * Ocols + c0]      = a10[r];
            Y2[row * Ocols + c0 + 16] = a11[r];
        }
    }
}

// ---------------------------------------------------------------------------
// BN statistics over (b,n,k) of hpre = Y1[idx] + Y2 (gather pass)
__global__ void k_edge_stats(const float* __restrict__ Y1, const float* __restrict__ Y2,
                             const int* __restrict__ idx, int O,
                             float* __restrict__ sum, float* __restrict__ sumsq) {
    __shared__ float ls[4][64], ls2[4][64];
    const int c = blockIdx.x * 64 + threadIdx.x;
    const int ty = threadIdx.y;
    float s = 0.f, s2 = 0.f;
    for (int row = blockIdx.y * 4 + ty; row < NROWS; row += 256) {
        int bbase = (row >> 10) << 10;
        float y2 = Y2[(size_t)row * O + c];
        const int* ir = idx + row * KNN;
#pragma unroll
        for (int k = 0; k < KNN; ++k) {
            float v = Y1[((size_t)bbase + ir[k]) * O + c] + y2;
            s += v; s2 += v * v;
        }
    }
    ls[ty][threadIdx.x] = s; ls2[ty][threadIdx.x] = s2;
    __syncthreads();
    if (ty == 0) {
        s  = ls[0][threadIdx.x] + ls[1][threadIdx.x] + ls[2][threadIdx.x] + ls[3][threadIdx.x];
        s2 = ls2[0][threadIdx.x] + ls2[1][threadIdx.x] + ls2[2][threadIdx.x] + ls2[3][threadIdx.x];
        atomicAdd(&sum[c], s); atomicAdd(&sumsq[c], s2);
    }
}

__global__ void k_finalize_bn(const float* __restrict__ sum, const float* __restrict__ sumsq,
                              const float* __restrict__ g, const float* __restrict__ bb,
                              float cnt, int O, float* __restrict__ scale, float* __restrict__ shift) {
    int o = blockIdx.x * 256 + threadIdx.x;
    if (o >= O) return;
    float m = sum[o] / cnt;
    float var = sumsq[o] / cnt - m * m;
    float inv = rsqrtf(var + 1e-5f);
    float sc = g[o] * inv;
    scale[o] = sc;
    shift[o] = bb[o] - m * sc;
}

// max over k of leaky(BN(Y1[idx]+Y2)), + optional residual; writes h_next and cat slice
__global__ void k_edge_max(const float* __restrict__ Y1, const float* __restrict__ Y2,
                           const int* __restrict__ idx,
                           const float* __restrict__ scale, const float* __restrict__ shift,
                           int O, const float* __restrict__ resid,
                           float* __restrict__ Hout, float* __restrict__ catp) {
    __shared__ int sidx[KNN];
    const int row = blockIdx.x;
    const int bbase = (row >> 10) << 10;
    if (threadIdx.x < KNN) sidx[threadIdx.x] = idx[row * KNN + threadIdx.x];
    __syncthreads();
    for (int o = threadIdx.x; o < O; o += 256) {
        float sc = scale[o], sh = shift[o];
        float y2 = Y2[(size_t)row * O + o];
        float best = -3.4e38f;
#pragma unroll
        for (int k = 0; k < KNN; ++k) {
            float v = (Y1[((size_t)bbase + sidx[k]) * O + o] + y2) * sc + sh;
            best = fmaxf(best, LRELU(v));
        }
        if (resid) best += resid[(size_t)row * O + o];
        Hout[(size_t)row * O + o] = best;
        catp[(size_t)row * CATCH + o] = best;
    }
}

// ---------------------------------------------------------------------------
// column stats for z (mean/var over all 4096 rows), O = 1024
__global__ void k_col_stats(const float* __restrict__ Z, int O,
                            float* __restrict__ sum, float* __restrict__ sumsq) {
    __shared__ float ls[4][64], ls2[4][64];
    const int c = blockIdx.x * 64 + threadIdx.x;
    const int ty = threadIdx.y;
    float s = 0.f, s2 = 0.f;
    for (int row = blockIdx.y * 4 + ty; row < NROWS; row += 64) {
        float v = Z[(size_t)row * O + c];
        s += v; s2 += v * v;
    }
    ls[ty][threadIdx.x] = s; ls2[ty][threadIdx.x] = s2;
    __syncthreads();
    if (ty == 0) {
        s  = ls[0][threadIdx.x] + ls[1][threadIdx.x] + ls[2][threadIdx.x] + ls[3][threadIdx.x];
        s2 = ls2[0][threadIdx.x] + ls2[1][threadIdx.x] + ls2[2][threadIdx.x] + ls2[3][threadIdx.x];
        atomicAdd(&sum[c], s); atomicAdd(&sumsq[c], s2);
    }
}

// BN + leaky + max/avg pool over the 1024 points: pooled (B, 2048) = [max | avg]
__global__ void k_bn_pool(const float* __restrict__ z, const float* __restrict__ scale,
                          const float* __restrict__ shift, float* __restrict__ pooled) {
    const int b = blockIdx.x;
    const int o = blockIdx.y * 256 + threadIdx.x;
    float sc = scale[o], sh = shift[o];
    float mx = -3.4e38f, s = 0.f;
    for (int n = 0; n < NPTS; ++n) {
        float v = z[((size_t)(b << 10) + n) * 1024 + o] * sc + sh;
        v = LRELU(v);
        mx = fmaxf(mx, v); s += v;
    }
    pooled[b * 2048 + o] = mx;
    pooled[b * 2048 + 1024 + o] = s * (1.f / 1024.f);
}

// ---------------------------------------------------------------------------
// tiny head: one thread per output channel does the 4-sample dot + BN(axis 0) + leaky
__global__ void k_head1(const float* __restrict__ pooled, const float* __restrict__ W,
                        const float* __restrict__ g, const float* __restrict__ bb,
                        float* __restrict__ z1) {
    int o = blockIdx.x * 256 + threadIdx.x;
    if (o >= 512) return;
    float a0 = 0, a1 = 0, a2 = 0, a3 = 0;
    for (int c = 0; c < 2048; ++c) {
        float w = W[(size_t)o * 2048 + c];
        a0 += pooled[c] * w; a1 += pooled[2048 + c] * w;
        a2 += pooled[4096 + c] * w; a3 += pooled[6144 + c] * w;
    }
    float m = 0.25f * (a0 + a1 + a2 + a3);
    float var = 0.25f * (a0 * a0 + a1 * a1 + a2 * a2 + a3 * a3) - m * m;
    float inv = rsqrtf(var + 1e-5f) * g[o];
    float sh = bb[o] - m * inv;
    float v;
    v = a0 * inv + sh; z1[0 * 512 + o] = LRELU(v);
    v = a1 * inv + sh; z1[1 * 512 + o] = LRELU(v);
    v = a2 * inv + sh; z1[2 * 512 + o] = LRELU(v);
    v = a3 * inv + sh; z1[3 * 512 + o] = LRELU(v);
}

__global__ void k_head2(const float* __restrict__ z1, const float* __restrict__ W,
                        const float* __restrict__ bias, const float* __restrict__ g,
                        const float* __restrict__ bb, float* __restrict__ z2) {
    int o = threadIdx.x;
    if (o >= 256) return;
    float a0 = bias[o], a1 = bias[o], a2 = bias[o], a3 = bias[o];
    for (int c = 0; c < 512; ++c) {
        float w = W[(size_t)o * 512 + c];
        a0 += z1[c] * w; a1 += z1[512 + c] * w;
        a2 += z1[1024 + c] * w; a3 += z1[1536 + c] * w;
    }
    float m = 0.25f * (a0 + a1 + a2 + a3);
    float var = 0.25f * (a0 * a0 + a1 * a1 + a2 * a2 + a3 * a3) - m * m;
    float inv = rsqrtf(var + 1e-5f) * g[o];
    float sh = bb[o] - m * inv;
    float v;
    v = a0 * inv + sh; z2[0 * 256 + o] = LRELU(v);
    v = a1 * inv + sh; z2[1 * 256 + o] = LRELU(v);
    v = a2 * inv + sh; z2[2 * 256 + o] = LRELU(v);
    v = a3 * inv + sh; z2[3 * 256 + o] = LRELU(v);
}

__global__ void k_head3(const float* __restrict__ z2, const float* __restrict__ W,
                        const float* __restrict__ bias, float* __restrict__ out) {
    int t = threadIdx.x;
    if (t >= 160) return;
    int s = t / 40, o = t % 40;
    float acc = bias[o];
    for (int c = 0; c < 256; ++c) acc += z2[s * 256 + c] * W[o * 256 + c];
    out[t] = acc;
}

// ---------------------------------------------------------------------------
extern "C" void kernel_launch(void* const* d_in, const int* in_sizes, int n_in,
                              void* d_out, int out_size, void* d_ws, size_t ws_size,
                              hipStream_t stream) {
    static const int IN_CH[14]  = {3, 64, 64, 128, 256, 256, 256, 256, 512, 512, 512, 512, 1024, 1024};
    static const int OUT_CH[14] = {64, 64, 128, 256, 256, 256, 256, 512, 512, 512, 512, 1024, 1024, 1024};
    const unsigned RES_MASK = (1u << 1) | (1u << 4) | (1u << 5) | (1u << 6) | (1u << 8) |
                              (1u << 9) | (1u << 10) | (1u << 12) | (1u << 13);

    const float* x = (const float*)d_in[0];
    const float* eW[14]; const float* eg[14]; const float* eb[14];
    for (int i = 0; i < 14; ++i) {
        eW[i] = (const float*)d_in[1 + i];
        eg[i] = (const float*)d_in[15 + i];
        eb[i] = (const float*)d_in[29 + i];
    }
    const float* W15  = (const float*)d_in[43];
    const float* g15  = (const float*)d_in[44];
    const float* b15  = (const float*)d_in[45];
    const float* W1   = (const float*)d_in[46];
    const float* g16  = (const float*)d_in[47];
    const float* b16  = (const float*)d_in[48];
    const float* W2   = (const float*)d_in[49];
    const float* bias2= (const float*)d_in[50];
    const float* g17  = (const float*)d_in[51];
    const float* b17  = (const float*)d_in[52];
    const float* W3   = (const float*)d_in[53];
    const float* bias3= (const float*)d_in[54];

    // workspace carve-up (floats)
    float* base = (float*)d_ws;
    size_t off = 0;
    float* hA    = base + off; off += (size_t)NROWS * 1024;
    float* hB    = base + off; off += (size_t)NROWS * 1024;
    float* dist  = base + off; off += (size_t)NROWS * NPTS;
    float* Y1    = base + off; off += (size_t)NROWS * 1024;
    float* Y2    = base + off; off += (size_t)NROWS * 1024;
    float* cat   = base + off; off += (size_t)NROWS * CATCH;
    float* z     = base + off; off += (size_t)NROWS * 1024;
    float* xx    = base + off; off += NROWS;
    int*   idx   = (int*)(base + off); off += NROWS * KNN;
    float* sum   = base + off; off += 1024;
    float* sumsq = base + off; off += 1024;
    float* scale = base + off; off += 1024;
    float* shift = base + off; off += 1024;
    float* pooled= base + off; off += BATCH * 2048;
    float* z1    = base + off; off += BATCH * 512;
    float* z2    = base + off; off += BATCH * 256;

    int catoff[14]; { int a = 0; for (int i = 0; i < 14; ++i) { catoff[i] = a; a += OUT_CH[i]; } }

    // h0 = transpose(x)
    k_transpose<<<dim3(NROWS / 256), dim3(256), 0, stream>>>(x, hA);

    float* hin = hA; float* hout = hB;
    for (int i = 0; i < 14; ++i) {
        const int C = IN_CH[i], O = OUT_CH[i];
        k_sqnorm<<<dim3(NROWS / 256), dim3(256), 0, stream>>>(hin, C, xx);
        if (C % 32 == 0)
            k_wmma_gram<true><<<dim3(NPTS / 64, NPTS / 64, BATCH), dim3(256), 0, stream>>>(hin, C, xx, dist);
        else
            k_wmma_gram<false><<<dim3(NPTS / 64, NPTS / 64, BATCH), dim3(256), 0, stream>>>(hin, C, xx, dist);
        k_topk<<<dim3(NROWS / 8), dim3(256), 0, stream>>>(dist, idx);
        if (C % 32 == 0)
            k_wmma_gemm_nt<1, true><<<dim3(NROWS / 64, O / 64), dim3(256), 0, stream>>>(hin, C, eW[i], 2 * C, Y1, Y2, O);
        else
            k_wmma_gemm_nt<1, false><<<dim3(NROWS / 64, O / 64), dim3(256), 0, stream>>>(hin, C, eW[i], 2 * C, Y1, Y2, O);
        hipMemsetAsync(sum, 0, 2 * 1024 * sizeof(float), stream);   // sum + sumsq contiguous
        k_edge_stats<<<dim3(O / 64, 64), dim3(64, 4), 0, stream>>>(Y1, Y2, idx, O, sum, sumsq);
        k_finalize_bn<<<dim3((O + 255) / 256), dim3(256), 0, stream>>>(sum, sumsq, eg[i], eb[i],
                                                                       (float)(NROWS * KNN), O, scale, shift);
        const float* resid = (RES_MASK >> i) & 1 ? hin : nullptr;
        k_edge_max<<<dim3(NROWS), dim3(256), 0, stream>>>(Y1, Y2, idx, scale, shift, O,
                                                          resid, hout, cat + catoff[i]);
        float* tmp = hin; hin = hout; hout = tmp;
    }

    // z = cat @ W15^T : (4096 x 6400) x (6400 x 1024)
    k_wmma_gemm_nt<0, true><<<dim3(NROWS / 64, 1024 / 64), dim3(256), 0, stream>>>(cat, CATCH, W15, CATCH, z, nullptr, 1024);
    hipMemsetAsync(sum, 0, 2 * 1024 * sizeof(float), stream);
    k_col_stats<<<dim3(16, 16), dim3(64, 4), 0, stream>>>(z, 1024, sum, sumsq);
    k_finalize_bn<<<dim3(4), dim3(256), 0, stream>>>(sum, sumsq, g15, b15, (float)NROWS, 1024, scale, shift);
    k_bn_pool<<<dim3(BATCH, 4), dim3(256), 0, stream>>>(z, scale, shift, pooled);

    k_head1<<<dim3(2), dim3(256), 0, stream>>>(pooled, W1, g16, b16, z1);
    k_head2<<<dim3(1), dim3(256), 0, stream>>>(z1, W2, bias2, g17, b17, z2);
    k_head3<<<dim3(1), dim3(192), 0, stream>>>(z2, W3, bias3, (float*)d_out);
}
